// MambaLayer_3178275799321
// MI455X (gfx1250) — matrix operand order repaired and checked by reading, not compiled, not run on hardware
//
#include <hip/hip_runtime.h>
#include <hip/hip_bf16.h>
#include <stdint.h>

// ---------------------------------------------------------------------------
// MambaLayer (PPM + selective scan) for MI455X / gfx1250, wave32, WMMA bf16.
// Big GEMMs (in_proj 215GF, out_proj 108GF, x_proj 9.5GF) use
// v_wmma_f32_16x16x32_bf16 with the shared 64xK B tile double-buffered in LDS
// (one cooperative VMEM fetch per block instead of 8 redundant per-wave
// streams); all 4 B fragments are loaded into independent registers before
// the WMMA burst so the LDS waits retire progressively. The sequential scan
// runs one thread per (b, d) channel with the 16-wide state in registers.
// ---------------------------------------------------------------------------

typedef __bf16 bf16_t;
typedef __attribute__((ext_vector_type(16))) __bf16 v16bf;
typedef __attribute__((ext_vector_type(8)))  __bf16 v8bf;
typedef __attribute__((ext_vector_type(8)))  float  v8f;

#define NB      64      // batch
#define LSEQ    256     // H*W
#define COMB    770
#define DMODEL  1282
#define DINNER  2564
#define DTRANK  81
#define NSTATE  16
#define MROWS   (NB*LSEQ)          // 16384
#define KPAD_IN  1312              // DMODEL padded to x32
#define NPAD_IN  5184              // 2*DINNER padded to x64
#define KPAD_U   2592              // DINNER padded to x32
#define NPAD_XP  128               // 113 -> 128
#define NPAD_OP  1344              // 1282 -> x64

#define LDS_KSTR 40                // LDS K-stride (elements): 80B rows ->
                                   // 16B-aligned frags, bank-conflict-free

// ============================ PPM stage ====================================

// seq[(b*256+l)*1282 + c] for c<770 : x | grid | pc_emb broadcast
__global__ void build_seq_comb(const float* __restrict__ x,
                               const float* __restrict__ pc,
                               float* __restrict__ seq) {
  int m = blockIdx.x;           // b*256 + l
  int b = m >> 8, l = m & 255;
  int h = l >> 4, w = l & 15;
  for (int c = threadIdx.x; c < COMB; c += 256) {
    float v;
    if (c < 512)        v = x[((size_t)b * 512 + c) * 256 + l];
    else if (c == 512)  v = -0.3f + 0.6f * (float)w / 15.0f;   // meshgrid X
    else if (c == 513)  v = -0.3f + 0.6f * (float)h / 15.0f;   // meshgrid Y
    else                v = pc[b * 256 + (c - 514)];
    seq[(size_t)m * DMODEL + c] = v;
  }
}

// pool scale 0: y0[b,d] = mean_l clip6(conv0(comb[b,l,:])) with folded BN
__global__ void pool0_kernel(const float* __restrict__ seq,
                             const float* __restrict__ cw,
                             const float* __restrict__ g,  const float* __restrict__ be,
                             const float* __restrict__ mu, const float* __restrict__ var,
                             float* __restrict__ y0) {
  __shared__ float sh[COMB];
  int b = blockIdx.x, d = threadIdx.x;          // block = 128
  float inv = rsqrtf(var[d] + 1e-5f) * g[d];
  float off = be[d] - mu[d] * inv;
  const float* wr = cw + (size_t)d * COMB;
  float acc = 0.0f;
  for (int l = 0; l < LSEQ; ++l) {
    const float* pr = seq + ((size_t)b * LSEQ + l) * DMODEL;
    for (int c = threadIdx.x; c < COMB; c += 128) sh[c] = pr[c];
    __syncthreads();
    float s = 0.0f;
    for (int c = 0; c < COMB; ++c) s += sh[c] * wr[c];
    s = s * inv + off;
    s = fminf(fmaxf(s, 0.0f), 6.0f);
    acc += s;
    __syncthreads();
  }
  y0[b * 128 + d] = acc * (1.0f / 256.0f);
}

__global__ void broadcast0(const float* __restrict__ y0, float* __restrict__ seq) {
  int m = blockIdx.x, b = m >> 8, d = threadIdx.x;     // block = 128
  seq[(size_t)m * DMODEL + COMB + d] = y0[b * 128 + d];
}

// adaptive avg pool over HxW bins -> pooled[b, rowOff + o*s+p, c]
__global__ void adaptive_pool(const float* __restrict__ seq, float* __restrict__ pooled,
                              int s, int rowOff) {
  int b = blockIdx.x, op = blockIdx.y;
  int o = op / s, p = op % s;
  int h0 = (o * 16) / s,  h1 = ((o + 1) * 16 + s - 1) / s;
  int w0 = (p * 16) / s,  w1 = ((p + 1) * 16 + s - 1) / s;
  float inv = 1.0f / (float)((h1 - h0) * (w1 - w0));
  for (int c = threadIdx.x; c < COMB; c += 256) {
    float acc = 0.0f;
    for (int h = h0; h < h1; ++h)
      for (int w = w0; w < w1; ++w)
        acc += seq[((size_t)b * LSEQ + (h * 16 + w)) * DMODEL + c];
    pooled[((size_t)b * 275 + rowOff + op) * COMB + c] = acc * inv;
  }
}

// 1x1 conv + folded BN + ReLU6 on pooled rows (LDS-staged input row)
__global__ void cbr_kernel(const float* __restrict__ pooled, const float* __restrict__ cw,
                           const float* __restrict__ g,  const float* __restrict__ be,
                           const float* __restrict__ mu, const float* __restrict__ var,
                           float* __restrict__ y, int rowOff) {
  __shared__ float sh[COMB];
  int b = blockIdx.x, row = rowOff + blockIdx.y, d = threadIdx.x;   // block = 128
  const float* pr = pooled + ((size_t)b * 275 + row) * COMB;
  for (int c = threadIdx.x; c < COMB; c += 128) sh[c] = pr[c];
  __syncthreads();
  const float* wr = cw + (size_t)d * COMB;
  float acc = 0.0f;
  for (int c = 0; c < COMB; ++c) acc += sh[c] * wr[c];
  float inv = rsqrtf(var[d] + 1e-5f) * g[d];
  float v = acc * inv + (be[d] - mu[d] * inv);
  v = fminf(fmaxf(v, 0.0f), 6.0f);
  y[((size_t)b * 275 + row) * 128 + d] = v;
}

// bilinear (align_corners=False) 16x16 upsample of s x s tile
__global__ void upsample_kernel(const float* __restrict__ y, float* __restrict__ seq,
                                int s, int rowOff, int chOff) {
  int b = blockIdx.x, l = blockIdx.y, d = threadIdx.x;              // block = 128
  int oh = l >> 4, ow = l & 15;
  auto tap = [&](int o, int& i0, int& i1, float& f) {
    float src = (o + 0.5f) * (float)s / 16.0f - 0.5f;
    src = fminf(fmaxf(src, 0.0f), (float)s - 1.0f);
    i0 = (int)floorf(src);
    i1 = min(i0 + 1, s - 1);
    f = src - (float)i0;
  };
  int h0, h1, w0, w1; float fh, fw;
  tap(oh, h0, h1, fh);
  tap(ow, w0, w1, fw);
  const float* yb = y + ((size_t)b * 275 + rowOff) * 128 + d;
  float v = (1.0f - fh) * (1.0f - fw) * yb[(h0 * s + w0) * 128]
          + (1.0f - fh) * fw          * yb[(h0 * s + w1) * 128]
          + fh          * (1.0f - fw) * yb[(h1 * s + w0) * 128]
          + fh          * fw          * yb[(h1 * s + w1) * 128];
  seq[((size_t)b * LSEQ + l) * DMODEL + chOff + d] = v;
}

// ============================ conversions ==================================

__global__ void cvt_seq_bf16(const float* __restrict__ seq, bf16_t* __restrict__ out) {
  int m = blockIdx.x;
  for (int c = threadIdx.x; c < KPAD_IN; c += 256)
    out[(size_t)m * KPAD_IN + c] =
        (c < DMODEL) ? (bf16_t)seq[(size_t)m * DMODEL + c] : (bf16_t)0.0f;
}

__global__ void cvt_w_bf16(const float* __restrict__ w, bf16_t* __restrict__ out,
                           int N, int K, int Kpad, size_t total) {
  size_t idx = (size_t)blockIdx.x * blockDim.x + threadIdx.x;
  if (idx >= total) return;
  int n = (int)(idx / Kpad), k = (int)(idx % Kpad);
  out[idx] = (n < N && k < K) ? (bf16_t)w[(size_t)n * K + k] : (bf16_t)0.0f;
}

__global__ void zero_u32(uint32_t* __restrict__ p, size_t n) {
  size_t i = (size_t)blockIdx.x * blockDim.x + threadIdx.x;
  if (i < n) p[i] = 0u;
}

// ============================ WMMA GEMM ====================================
// C(M,N) = A(M,Kpad) * W(N,Kpad)^T. Block = 8 waves stacked in M (128x64
// block tile); the 64x32 B tile shared by all waves is staged in LDS
// (double-buffered, one barrier/k-step, 80B row pitch -> 16B-aligned,
// bank-conflict-free ds_load_b128 fragment reads). All 4 B fragments load
// into independent registers before the 4-WMMA burst so dscnt waits retire
// progressively. A fragments stream from VMEM with prefetch; f32 accumulate.
__global__ __launch_bounds__(256) void wmma_gemm(
    const bf16_t* __restrict__ A, const bf16_t* __restrict__ W,
    float* __restrict__ C, int ldk, int kSteps, int Ncap, int ldc,
    int transpose_out) {
  __shared__ bf16_t sB[2][64 * LDS_KSTR];     // 2 x 5 KB

  const int lane = threadIdx.x & 31;
  const int wave = threadIdx.x >> 5;
  const int tileM = (blockIdx.y * 8 + wave) * 16;
  const int tileN = blockIdx.x * 64;

  // A fragment addressing (16x32 bf16, ISA 7.12.2): lane -> row m, K halves
  const int mRow = tileM + (lane & 15);
  const int aKb  = (lane & 16) ? 8 : 0;
  // B fragment addressing (32x16 bf16): lane -> column n, K halves
  const int nCol = lane & 15;
  const int bKb  = (lane & 16) ? 8 : 0;

  // cooperative B-tile fetch mapping: 64 rows x 32 K; 8 bf16 (16B) per thread
  const int cRow = threadIdx.x >> 2;
  const int cKo  = (threadIdx.x & 3) * 8;
  const bf16_t* wSrc = W + (size_t)(tileN + cRow) * ldk + cKo;

  v8f acc[4];
#pragma unroll
  for (int j = 0; j < 4; ++j)
#pragma unroll
    for (int r = 0; r < 8; ++r) acc[j][r] = 0.0f;

  const bf16_t* aBase = A + (size_t)mRow * ldk + aKb;

  // prime buffer 0
  *(v8bf*)&sB[0][cRow * LDS_KSTR + cKo] = *(const v8bf*)(wSrc);
  __syncthreads();

  int cur = 0;
  for (int ks = 0; ks < kSteps; ++ks) {
    // issue next B tile fetch from VMEM while computing on current buffer
    v8bf nextB;
    const bool hasNext = (ks + 1 < kSteps);
    if (hasNext) nextB = *(const v8bf*)(wSrc + (ks + 1) * 32);

    union { v16bf v; v8bf h[2]; } af;
    const bf16_t* ap = aBase + ks * 32;
    af.h[0] = *(const v8bf*)(ap);        // K = kb .. kb+7
    af.h[1] = *(const v8bf*)(ap + 16);   // K = kb+16 .. kb+23
    __builtin_prefetch(ap + 32, 0, 1);   // next K-tile of A (global_prefetch_b8)

    // load all 4 B fragments first (independent regs -> progressive dscnt)
    union { v16bf v; v8bf h[2]; } bfr[4];
    const bf16_t* bl = &sB[cur][nCol * LDS_KSTR + bKb];
#pragma unroll
    for (int j = 0; j < 4; ++j) {
      const bf16_t* bp = bl + j * 16 * LDS_KSTR;
      bfr[j].h[0] = *(const v8bf*)(bp);  // ds_load_b128
      bfr[j].h[1] = *(const v8bf*)(bp + 16);
    }
#pragma unroll
    for (int j = 0; j < 4; ++j)
      acc[j] = __builtin_amdgcn_wmma_f32_16x16x32_bf16(
          false, af.v, false, bfr[j].v, (short)0, acc[j], false, false);

    if (hasNext) {
      *(v8bf*)&sB[cur ^ 1][cRow * LDS_KSTR + cKo] = nextB;
      __syncthreads();                   // stores visible + everyone done with cur
      cur ^= 1;
    }
  }

  const int mBase = tileM + ((lane & 16) ? 8 : 0);
#pragma unroll
  for (int j = 0; j < 4; ++j) {
    int n = tileN + j * 16 + nCol;
    if (n >= Ncap) continue;
#pragma unroll
    for (int r = 0; r < 8; ++r) {
      int m = mBase + r;
      if (!transpose_out) {
        C[(size_t)m * ldc + n] = acc[j][r];
      } else {               // store as (b, n, l): final output layout (B,C,H,W)
        int bb = m >> 8, ll = m & 255;
        C[((size_t)bb * Ncap + n) * 256 + ll] = acc[j][r];
      }
    }
  }
}

// ============================ Mamba stage ==================================

// causal depthwise conv1d (+bias) + SiLU; writes f32 u and padded bf16 u
__global__ void conv1d_silu(const float* __restrict__ xz,
                            const float* __restrict__ cw, const float* __restrict__ cb,
                            float* __restrict__ u, bf16_t* __restrict__ ub) {
  int m = blockIdx.x;
  int d = blockIdx.y * 256 + threadIdx.x;
  if (d >= KPAD_U) return;
  if (d >= DINNER) { ub[(size_t)m * KPAD_U + d] = (bf16_t)0.0f; return; }
  int l = m & 255;
  float acc = cb[d];
#pragma unroll
  for (int j = 0; j < 4; ++j) {
    int ll = l - 3 + j;
    if (ll >= 0) acc += cw[d * 4 + j] * xz[(size_t)(m - 3 + j) * (2 * DINNER) + d];
  }
  float s = acc / (1.0f + __expf(-acc));  // SiLU
  u[(size_t)m * DINNER + d] = s;
  ub[(size_t)m * KPAD_U + d] = (bf16_t)s;
}

// dt = softplus(dbl[:, :81] @ dt_proj_w^T + b)
__global__ void dtproj_softplus(const float* __restrict__ dbl,
                                const float* __restrict__ dw, const float* __restrict__ db,
                                float* __restrict__ dt) {
  __shared__ float sh[DTRANK];
  int m = blockIdx.x;
  int d = blockIdx.y * 256 + threadIdx.x;
  if (threadIdx.x < DTRANK) sh[threadIdx.x] = dbl[(size_t)m * 113 + threadIdx.x];
  __syncthreads();
  if (d >= DINNER) return;
  const float* wr = dw + (size_t)d * DTRANK;
  float acc = db[d];
  for (int r = 0; r < DTRANK; ++r) acc += sh[r] * wr[r];
  float v = (acc > 20.0f) ? acc : log1pf(__expf(acc));
  dt[(size_t)m * DINNER + d] = v;
}

// selective scan: h[16] in registers, B/C staged in LDS per step; fused
// +u*D and SiLU(z) gate; emits bf16 rows for the out_proj WMMA GEMM.
__global__ void scan_kernel(const float* __restrict__ dt, const float* __restrict__ u,
                            const float* __restrict__ dbl, const float* __restrict__ xz,
                            const float* __restrict__ A_log, const float* __restrict__ Dp,
                            bf16_t* __restrict__ yb) {
  __shared__ float sB[NSTATE], sC[NSTATE];
  int b = blockIdx.x;
  int d = blockIdx.y * 256 + threadIdx.x;
  bool act = (d < DINNER);
  float A[NSTATE], h[NSTATE], Dd = 0.0f;
  if (act) {
#pragma unroll
    for (int n = 0; n < NSTATE; ++n) {
      A[n] = -__expf(A_log[(size_t)d * NSTATE + n]);
      h[n] = 0.0f;
    }
    Dd = Dp[d];
  }
  for (int l = 0; l < LSEQ; ++l) {
    size_t m = (size_t)b * LSEQ + l;
    __syncthreads();
    if (threadIdx.x < 32) {
      float v = dbl[m * 113 + DTRANK + threadIdx.x];
      if (threadIdx.x < 16) sB[threadIdx.x] = v;
      else                  sC[threadIdx.x - 16] = v;
    }
    __syncthreads();
    if (act) {
      float dtv = dt[m * DINNER + d];
      float uv  = u[m * DINNER + d];
      float du  = dtv * uv;
      float y = 0.0f;
#pragma unroll
      for (int n = 0; n < NSTATE; ++n) {
        h[n] = __expf(dtv * A[n]) * h[n] + du * sB[n];
        y += h[n] * sC[n];
      }
      float z = xz[m * (2 * DINNER) + DINNER + d];
      float out = (y + uv * Dd) * (z / (1.0f + __expf(-z)));
      yb[m * KPAD_U + d] = (bf16_t)out;
    }
  }
}

// ============================ orchestration ================================

extern "C" void kernel_launch(void* const* d_in, const int* in_sizes, int n_in,
                              void* d_out, int out_size, void* d_ws, size_t ws_size,
                              hipStream_t stream) {
  const float* x       = (const float*)d_in[0];
  const float* pc      = (const float*)d_in[1];
  const float* conv_w  = (const float*)d_in[2];
  const float* bn_g    = (const float*)d_in[3];
  const float* bn_b    = (const float*)d_in[4];
  const float* bn_m    = (const float*)d_in[5];
  const float* bn_v    = (const float*)d_in[6];
  const float* in_w    = (const float*)d_in[7];
  const float* c1w     = (const float*)d_in[8];
  const float* c1b     = (const float*)d_in[9];
  const float* xpw     = (const float*)d_in[10];
  const float* dtw     = (const float*)d_in[11];
  const float* dtb     = (const float*)d_in[12];
  const float* A_log   = (const float*)d_in[13];
  const float* Dp      = (const float*)d_in[14];
  const float* opw     = (const float*)d_in[15];
  float* out = (float*)d_out;

  char* ws = (char*)d_ws;
  size_t off = 0;
  auto alloc = [&](size_t bytes) -> char* {
    char* p = ws + off;
    off += (bytes + 255) & ~(size_t)255;
    return p;
  };
  float*  seq      = (float*) alloc((size_t)MROWS * DMODEL * 4);
  bf16_t* seq_bf   = (bf16_t*)alloc((size_t)MROWS * KPAD_IN * 2);
  bf16_t* w_in_bf  = (bf16_t*)alloc((size_t)NPAD_IN * KPAD_IN * 2);
  float*  pooled   = (float*) alloc((size_t)NB * 275 * COMB * 4);
  float*  y_cbr    = (float*) alloc((size_t)NB * 275 * 128 * 4);
  float*  y0       = (float*) alloc((size_t)NB * 128 * 4);
  float*  xz       = (float*) alloc((size_t)MROWS * 2 * DINNER * 4);
  float*  u        = (float*) alloc((size_t)MROWS * DINNER * 4);
  bf16_t* u_bf     = (bf16_t*)alloc((size_t)MROWS * KPAD_U * 2);
  bf16_t* xpw_bf   = (bf16_t*)alloc((size_t)NPAD_XP * KPAD_U * 2);
  float*  dbl      = (float*) alloc((size_t)MROWS * 113 * 4);
  float*  dt       = (float*) alloc((size_t)MROWS * DINNER * 4);
  bf16_t* y_bf     = (bf16_t*)alloc((size_t)MROWS * KPAD_U * 2);
  bf16_t* opw_bf   = (bf16_t*)alloc((size_t)NPAD_OP * KPAD_U * 2);
  (void)ws_size; (void)in_sizes; (void)n_in; (void)out_size;

  const int scales[3]  = {5, 9, 13};
  const int rowOffs[3] = {0, 25, 106};   // 25 + 81 + 169 = 275 rows

  // --- PPM ---
  build_seq_comb<<<MROWS, 256, 0, stream>>>(x, pc, seq);
  pool0_kernel<<<NB, 128, 0, stream>>>(seq, conv_w, bn_g, bn_b, bn_m, bn_v, y0);
  broadcast0<<<MROWS, 128, 0, stream>>>(y0, seq);
  for (int i = 0; i < 3; ++i) {
    int s = scales[i], ro = rowOffs[i];
    adaptive_pool<<<dim3(NB, s * s), 256, 0, stream>>>(seq, pooled, s, ro);
    cbr_kernel<<<dim3(NB, s * s), 128, 0, stream>>>(
        pooled, conv_w + (size_t)(i + 1) * 128 * COMB,
        bn_g + (i + 1) * 128, bn_b + (i + 1) * 128,
        bn_m + (i + 1) * 128, bn_v + (i + 1) * 128, y_cbr, ro);
    upsample_kernel<<<dim3(NB, LSEQ), 128, 0, stream>>>(
        y_cbr, seq, s, ro, COMB + 128 * (i + 1));
  }

  // --- in_proj (WMMA) ---
  cvt_seq_bf16<<<MROWS, 256, 0, stream>>>(seq, seq_bf);
  {
    size_t tot = (size_t)NPAD_IN * KPAD_IN;
    cvt_w_bf16<<<(unsigned)((tot + 255) / 256), 256, 0, stream>>>(
        in_w, w_in_bf, 2 * DINNER, DMODEL, KPAD_IN, tot);
  }
  wmma_gemm<<<dim3(NPAD_IN / 64, MROWS / 128), 256, 0, stream>>>(
      seq_bf, w_in_bf, xz, KPAD_IN, KPAD_IN / 32, 2 * DINNER, 2 * DINNER, 0);

  // --- conv1d + SiLU ---
  conv1d_silu<<<dim3(MROWS, (KPAD_U + 255) / 256), 256, 0, stream>>>(xz, c1w, c1b, u, u_bf);

  // --- x_proj (WMMA) ---
  {
    size_t tot = (size_t)NPAD_XP * KPAD_U;
    cvt_w_bf16<<<(unsigned)((tot + 255) / 256), 256, 0, stream>>>(
        xpw, xpw_bf, 113, DINNER, KPAD_U, tot);
  }
  wmma_gemm<<<dim3(NPAD_XP / 64, MROWS / 128), 256, 0, stream>>>(
      u_bf, xpw_bf, dbl, KPAD_U, KPAD_U / 32, 113, 113, 0);

  // --- dt_proj + softplus ---
  dtproj_softplus<<<dim3(MROWS, (DINNER + 255) / 256), 256, 0, stream>>>(dbl, dtw, dtb, dt);

  // --- selective scan (emits bf16 rows, pads pre-zeroed) ---
  {
    size_t nwords = (size_t)MROWS * KPAD_U * 2 / 4;
    zero_u32<<<(unsigned)((nwords + 255) / 256), 256, 0, stream>>>((uint32_t*)y_bf, nwords);
  }
  scan_kernel<<<dim3(NB, (DINNER + 255) / 256), 256, 0, stream>>>(
      dt, u, dbl, xz, A_log, Dp, y_bf);

  // --- out_proj (WMMA, fused transposed store -> (B, C, H, W)) ---
  {
    size_t tot = (size_t)NPAD_OP * KPAD_U;
    cvt_w_bf16<<<(unsigned)((tot + 255) / 256), 256, 0, stream>>>(
        opw, opw_bf, DMODEL, DINNER, KPAD_U, tot);
  }
  wmma_gemm<<<dim3(NPAD_OP / 64, MROWS / 128), 256, 0, stream>>>(
      y_bf, opw_bf, out, KPAD_U, KPAD_U / 32, DMODEL, DMODEL, 1);
}
